// LLMConditioner_89034672046278
// MI455X (gfx1250) — compile-verified, matrix operand order checked by reading, not run on hardware
//
#include <hip/hip_runtime.h>
#include <hip/hip_bf16.h>
#include <math.h>

// ---- problem dims ----
#define BB 4
#define PREFIX 16
#define PROMPT 3
#define LAB 61
#define TT 80          // PREFIX+PROMPT+LAB
#define DD 2048
#define NL 2
#define NH 32
#define NKV 4
#define HD 64
#define FF 5632
#define VV 32000
#define RR 4
#define MROWS (BB*TT)  // 320

typedef __attribute__((ext_vector_type(16))) _Float16 v16h;
typedef __attribute__((ext_vector_type(8)))  float    v8f;

__device__ __constant__ float NF4_LUT[16] = {
  -1.0f, -0.6961928f, -0.52507305f, -0.39491748f, -0.28444138f, -0.18477343f,
  -0.09105003f, 0.0f, 0.0795803f, 0.1609302f, 0.2461123f, 0.33791524f,
  0.44070983f, 0.562617f, 0.72295684f, 1.0f };

// ISA 16-bit A/B 16x32 striping: lanes 0-15 hold K={0..7,16..23} pairs,
// lanes 16-31 hold K={8..15,24..31} pairs.  Per lane the 16 elements are two
// contiguous 8-element runs at k0+8*half and k0+16+8*half.

__device__ __forceinline__ void load_a_frag(const float* __restrict__ xrow,
                                            int k0, int half, v16h& a)
{
  const float4 a0 = *(const float4*)(xrow + k0 + 8 * half);
  const float4 a1 = *(const float4*)(xrow + k0 + 8 * half + 4);
  const float4 a2 = *(const float4*)(xrow + k0 + 16 + 8 * half);
  const float4 a3 = *(const float4*)(xrow + k0 + 16 + 8 * half + 4);
  a[0]  = (_Float16)a0.x;  a[1]  = (_Float16)a0.y;
  a[2]  = (_Float16)a0.z;  a[3]  = (_Float16)a0.w;
  a[4]  = (_Float16)a1.x;  a[5]  = (_Float16)a1.y;
  a[6]  = (_Float16)a1.z;  a[7]  = (_Float16)a1.w;
  a[8]  = (_Float16)a2.x;  a[9]  = (_Float16)a2.y;
  a[10] = (_Float16)a2.z;  a[11] = (_Float16)a2.w;
  a[12] = (_Float16)a3.x;  a[13] = (_Float16)a3.y;
  a[14] = (_Float16)a3.z;  a[15] = (_Float16)a3.w;
}

// dequant 4 int32 NF4 codes -> 4 f16 lanes of fragment (shuffle-based LUT)
template<int BASE>
__device__ __forceinline__ void dq4(const int4 cw, float lutv, float am,
                                    v16h& b)
{
  b[BASE + 0] = (_Float16)(__shfl(lutv, cw.x & 15, 32) * am);
  b[BASE + 1] = (_Float16)(__shfl(lutv, cw.y & 15, 32) * am);
  b[BASE + 2] = (_Float16)(__shfl(lutv, cw.z & 15, 32) * am);
  b[BASE + 3] = (_Float16)(__shfl(lutv, cw.w & 15, 32) * am);
}

// ---------------- WMMA GEMM, NF4-dequant weights ----------------
// Y[M,N] (+)= X[M,K] * W^T, codes[N,K] int32 0..15, absmax per 64 along K.
// One wave -> one 16-row M tile x NT 16-col N tiles (A reused across tiles).
template<int NT>
__global__ void gemm_nf4_kernel(const float* __restrict__ X, int M, int K,
                                const int* __restrict__ codes,
                                const float* __restrict__ absmax,
                                int N, float* __restrict__ Y, int ldy,
                                int accumulate)
{
  const int lane = threadIdx.x & 31;
  const int wv = (int)((blockIdx.x * blockDim.x + threadIdx.x) >> 5);
  const int ngroups = N / (16 * NT);
  const int mtiles = M >> 4;           // M % 16 == 0 for all callers
  if (wv >= ngroups * mtiles) return;  // wave-uniform
  const int mt = wv / ngroups, ng = wv % ngroups;
  const int m0 = mt * 16, n0 = ng * (16 * NT);
  const int half = lane >> 4, idx = lane & 15;
  const float* xrow = X + (size_t)(m0 + idx) * K;
  const int kb = K >> 6;               // absmax blocks per row
  const int* wrow0 = codes + (size_t)(n0 + idx) * K;
  const float* am0 = absmax + (size_t)(n0 + idx) * kb;
  const float lutv = NF4_LUT[lane & 15];

  v8f c[NT];
#pragma unroll
  for (int j = 0; j < NT; ++j) c[j] = {};

  for (int k0 = 0; k0 < K; k0 += 32) {
    v16h a;
    load_a_frag(xrow, k0, half, a);
#pragma unroll
    for (int j = 0; j < NT; ++j) {
      const int* wr = wrow0 + (size_t)j * 16 * K;
      __builtin_prefetch((const void*)(wr + k0 + 256), 0, 1);
      const int4 w0 = *(const int4*)(wr + k0 + 8 * half);
      const int4 w1 = *(const int4*)(wr + k0 + 8 * half + 4);
      const int4 w2 = *(const int4*)(wr + k0 + 16 + 8 * half);
      const int4 w3 = *(const int4*)(wr + k0 + 16 + 8 * half + 4);
      const float am = am0[(size_t)j * 16 * kb + (k0 >> 6)];
      v16h b;
      dq4<0>(w0, lutv, am, b);
      dq4<4>(w1, lutv, am, b);
      dq4<8>(w2, lutv, am, b);
      dq4<12>(w3, lutv, am, b);
      c[j] = __builtin_amdgcn_wmma_f32_16x16x32_f16(false, a, false, b,
                                                    (short)0, c[j],
                                                    false, false);
    }
  }
#pragma unroll
  for (int j = 0; j < NT; ++j) {
#pragma unroll
    for (int r = 0; r < 8; ++r) {
      const size_t off = (size_t)(m0 + r + 8 * half) * ldy + n0 + j * 16 + idx;
      if (accumulate) Y[off] += c[j][r]; else Y[off] = c[j][r];
    }
  }
}

// ---------------- WMMA GEMM, fp32 weights (K-major) ----------------
// Y[M,N] = X[M,K] * W[K,N] (+ bias). M % 16 == 0 (conditioner is padded);
// only the first Mstore rows are written.
template<int NT>
__global__ void gemm_f32_kernel(const float* __restrict__ X, int M, int K,
                                const float* __restrict__ W, int ldw,
                                const float* __restrict__ bias,
                                int N, float* __restrict__ Y, int ldy,
                                int Mstore)
{
  const int lane = threadIdx.x & 31;
  const int wv = (int)((blockIdx.x * blockDim.x + threadIdx.x) >> 5);
  const int ngroups = N / (16 * NT);
  const int mtiles = M >> 4;
  if (wv >= ngroups * mtiles) return;
  const int mt = wv / ngroups, ng = wv % ngroups;
  const int m0 = mt * 16, n0 = ng * (16 * NT);
  const int half = lane >> 4, idx = lane & 15;
  const float* xrow = X + (size_t)(m0 + idx) * K;

  v8f c[NT];
#pragma unroll
  for (int j = 0; j < NT; ++j) c[j] = {};

  for (int k0 = 0; k0 < K; k0 += 32) {
    v16h a;
    load_a_frag(xrow, k0, half, a);
#pragma unroll
    for (int j = 0; j < NT; ++j) {
      const int ncol = n0 + j * 16 + idx;
      v16h b;
#pragma unroll
      for (int v = 0; v < 8; ++v) {
        const int k = k0 + ((v < 4) ? 2 * v : 2 * v + 8) + 8 * half;
        b[2 * v]     = (_Float16)W[(size_t)k * ldw + ncol];
        b[2 * v + 1] = (_Float16)W[(size_t)(k + 1) * ldw + ncol];
      }
      c[j] = __builtin_amdgcn_wmma_f32_16x16x32_f16(false, a, false, b,
                                                    (short)0, c[j],
                                                    false, false);
    }
  }
#pragma unroll
  for (int j = 0; j < NT; ++j) {
    const float bv = bias ? bias[n0 + j * 16 + idx] : 0.f;
#pragma unroll
    for (int r = 0; r < 8; ++r) {
      const int m = m0 + r + 8 * half;
      if (m < Mstore)
        Y[(size_t)m * ldy + n0 + j * 16 + idx] = c[j][r] + bv;
    }
  }
}

// ---------------- small scalar kernels ----------------
__global__ void set_zero_kernel(float* p) { p[0] = 0.f; }

// writes 16 rows: rows 0..3 = gelu(s_bt@w1+b1), rows 4..15 = 0 (padding)
__global__ void cond_h1_kernel(const float* __restrict__ s_bt,
                               const float* __restrict__ w1,
                               const float* __restrict__ b1,
                               float* __restrict__ h1)
{
  const int i = blockIdx.x * blockDim.x + threadIdx.x;
  if (i >= 16 * 4 * DD) return;
  const int m = i >> 13, n = i & (4 * DD - 1);
  if (m >= BB) { h1[i] = 0.f; return; }
  float s = b1[n];
  const float* xr = s_bt + m * 512;
  for (int k = 0; k < 512; ++k) s += xr[k] * w1[(size_t)k * (4 * DD) + n];
  h1[i] = 0.5f * s * (1.0f + erff(s * 0.70710678f));   // exact gelu
}

__global__ void embed_gather_kernel(const float* __restrict__ embed,
                                    const int* __restrict__ pids,
                                    const int* __restrict__ lids,
                                    float* __restrict__ x)
{
  const int t = blockIdx.x;            // 0..63 -> slot PREFIX+t
  const int b = blockIdx.y;
  const int id = (t < PROMPT) ? pids[b * PROMPT + t] : lids[b * LAB + (t - PROMPT)];
  const float* src = embed + (size_t)id * DD;
  float* dst = x + ((size_t)(b * TT + PREFIX + t)) * DD;
  for (int i = threadIdx.x; i < DD; i += blockDim.x) dst[i] = src[i];
}

__device__ __forceinline__ float block_reduce_sum(float v, float* sm) {
  const int tid = threadIdx.x;
  sm[tid] = v; __syncthreads();
  for (int s = blockDim.x >> 1; s > 0; s >>= 1) {
    if (tid < s) sm[tid] += sm[tid + s];
    __syncthreads();
  }
  const float r = sm[0]; __syncthreads();
  return r;
}

__device__ __forceinline__ float block_reduce_max(float v, float* sm) {
  const int tid = threadIdx.x;
  sm[tid] = v; __syncthreads();
  for (int s = blockDim.x >> 1; s > 0; s >>= 1) {
    if (tid < s) sm[tid] = fmaxf(sm[tid], sm[tid + s]);
    __syncthreads();
  }
  const float r = sm[0]; __syncthreads();
  return r;
}

__global__ void rmsnorm_kernel(const float* __restrict__ x,
                               const float* __restrict__ g,
                               float* __restrict__ y)
{
  __shared__ float sm[256];
  const int row = blockIdx.x;
  const float* xr = x + (size_t)row * DD;
  float ss = 0.f;
  for (int i = threadIdx.x; i < DD; i += blockDim.x) { float v = xr[i]; ss += v * v; }
  const float tot = block_reduce_sum(ss, sm);
  const float inv = rsqrtf(tot / (float)DD + 1e-5f);
  float* yr = y + (size_t)row * DD;
  for (int i = threadIdx.x; i < DD; i += blockDim.x) yr[i] = xr[i] * inv * g[i];
}

__global__ void lora_a_kernel(const float* __restrict__ hn,
                              const float* __restrict__ lA,
                              float* __restrict__ t1)
{
  const int i = blockIdx.x * blockDim.x + threadIdx.x;
  if (i >= MROWS * RR) return;
  const int row = i >> 2, r = i & 3;
  const float* xr = hn + (size_t)row * DD;
  const float* ar = lA + (size_t)r * DD;
  float s = 0.f;
  for (int k = 0; k < DD; ++k) s += xr[k] * ar[k];
  t1[i] = s;
}

__global__ void lora_b_add_kernel(const float* __restrict__ t1,
                                  const float* __restrict__ lB,
                                  float* __restrict__ q)
{
  const int i = blockIdx.x * blockDim.x + threadIdx.x;
  if (i >= MROWS * DD) return;
  const int row = i >> 11, n = i & (DD - 1);
  float s = 0.f;
#pragma unroll
  for (int r = 0; r < RR; ++r) s += t1[row * RR + r] * lB[(size_t)n * RR + r];
  q[i] += 4.0f * s;   // ALPHA/R = 16/4
}

__global__ void rope_kernel(float* __restrict__ buf, int nheads)
{
  const int i = blockIdx.x * blockDim.x + threadIdx.x;
  const int total = MROWS * nheads * (HD / 2);
  if (i >= total) return;
  const int d = i & 31;
  const int h = (i >> 5) % nheads;
  const int row = i / (32 * nheads);
  const int t = row % TT;
  const float inv = powf(10000.0f, -(float)d / 32.0f);
  const float ang = (float)t * inv;
  const float cc = cosf(ang), ssn = sinf(ang);
  const size_t base = (size_t)row * (nheads * HD) + h * HD;
  const float t1 = buf[base + d], t2 = buf[base + 32 + d];
  buf[base + d]      = t1 * cc - t2 * ssn;
  buf[base + 32 + d] = t2 * cc + t1 * ssn;
}

__global__ void attn_kernel(const float* __restrict__ q,
                            const float* __restrict__ k,
                            const float* __restrict__ v,
                            float* __restrict__ o)
{
  __shared__ float Ks[TT][HD];
  __shared__ float Vs[TT][HD];
  const int b = blockIdx.x, h = blockIdx.y;
  const int kvh = h >> 3;                // NH/NKV = 8
  for (int i = threadIdx.x; i < TT * HD; i += blockDim.x) {
    const int t = i >> 6, d = i & 63;
    Ks[t][d] = k[(size_t)(b * TT + t) * (NKV * HD) + kvh * HD + d];
    Vs[t][d] = v[(size_t)(b * TT + t) * (NKV * HD) + kvh * HD + d];
  }
  __syncthreads();
  for (int tq = threadIdx.x; tq < TT; tq += blockDim.x) {
    const float* qr = q + (size_t)(b * TT + tq) * DD + h * HD;
    float qreg[HD];
#pragma unroll
    for (int d = 0; d < HD; ++d) qreg[d] = qr[d];
    float m = -1e30f, sum = 0.f;
    float acc[HD];
#pragma unroll
    for (int d = 0; d < HD; ++d) acc[d] = 0.f;
    for (int tk = 0; tk <= tq; ++tk) {
      float s = 0.f;
#pragma unroll
      for (int d = 0; d < HD; ++d) s += qreg[d] * Ks[tk][d];
      s *= 0.125f;                        // 1/sqrt(64)
      if (s > m) {
        const float sc = expf(m - s);
        sum *= sc;
#pragma unroll
        for (int d = 0; d < HD; ++d) acc[d] *= sc;
        m = s;
      }
      const float w = expf(s - m);
      sum += w;
#pragma unroll
      for (int d = 0; d < HD; ++d) acc[d] += w * Vs[tk][d];
    }
    float* orow = o + (size_t)(b * TT + tq) * DD + h * HD;
    const float rinv = 1.0f / sum;
#pragma unroll
    for (int d = 0; d < HD; ++d) orow[d] = acc[d] * rinv;
  }
}

__global__ void silu_mul_kernel(const float* __restrict__ g,
                                const float* __restrict__ u,
                                float* __restrict__ f)
{
  const int i = blockIdx.x * blockDim.x + threadIdx.x;
  if (i >= MROWS * FF) return;
  const float gv = g[i];
  f[i] = (gv / (1.0f + expf(-gv))) * u[i];
}

__global__ void nll_kernel(const float* __restrict__ logits,
                           const int* __restrict__ lids,
                           float* __restrict__ out)
{
  __shared__ float sm[256];
  const int ti = blockIdx.x;             // 0..60
  const int b = blockIdx.y;
  const int row = b * TT + 18 + ti;      // sl position t = 18..78
  const int target = lids[b * LAB + ti]; // labels[b, t+1] = label_ids[b, t-18]
  const float* lr = logits + (size_t)row * VV;
  float mx = -1e30f;
  for (int vi = threadIdx.x; vi < VV; vi += blockDim.x) mx = fmaxf(mx, lr[vi]);
  mx = block_reduce_max(mx, sm);
  float se = 0.f;
  for (int vi = threadIdx.x; vi < VV; vi += blockDim.x) se += expf(lr[vi] - mx);
  se = block_reduce_sum(se, sm);
  if (threadIdx.x == 0) {
    const float nll = (logf(se) + mx) - lr[target];
    atomicAdd(out, nll * (1.0f / (float)(BB * LAB)));   // /244
  }
}

// ---------------- host ----------------
#define NTILE 4
static inline dim3 gemm_grid(int M, int N) {
  long waves = (long)(M / 16) * (long)(N / (16 * NTILE));
  long thr = waves * 32;
  return dim3((unsigned)((thr + 255) / 256));
}

extern "C" void kernel_launch(void* const* d_in, const int* in_sizes, int n_in,
                              void* d_out, int out_size, void* d_ws, size_t ws_size,
                              hipStream_t stream)
{
  const float* s_bt       = (const float*)d_in[0];
  const int*   prompt_ids = (const int*)  d_in[1];
  const int*   label_ids  = (const int*)  d_in[2];
  const float* embed      = (const float*)d_in[3];
  const float* w1         = (const float*)d_in[4];
  const float* b1         = (const float*)d_in[5];
  const float* w2         = (const float*)d_in[6];
  const float* b2         = (const float*)d_in[7];
  const float* loraA      = (const float*)d_in[8];
  const float* loraB      = (const float*)d_in[9];
  const int*   q_codes    = (const int*)  d_in[10];
  const float* q_absmax   = (const float*)d_in[11];
  const int*   k_codes    = (const int*)  d_in[12];
  const float* k_absmax   = (const float*)d_in[13];
  const int*   v_codes    = (const int*)  d_in[14];
  const float* v_absmax   = (const float*)d_in[15];
  const int*   o_codes    = (const int*)  d_in[16];
  const float* o_absmax   = (const float*)d_in[17];
  const int*   g_codes    = (const int*)  d_in[18];
  const float* g_absmax   = (const float*)d_in[19];
  const int*   u_codes    = (const int*)  d_in[20];
  const float* u_absmax   = (const float*)d_in[21];
  const int*   d_codes    = (const int*)  d_in[22];
  const float* d_absmax   = (const float*)d_in[23];
  const float* attn_norm  = (const float*)d_in[24];
  const float* mlp_norm   = (const float*)d_in[25];
  const float* final_norm = (const float*)d_in[26];
  const float* lm_head    = (const float*)d_in[27];
  float* out = (float*)d_out;
  (void)in_sizes; (void)n_in; (void)out_size; (void)ws_size;

  // workspace layout (floats)
  float* ws = (float*)d_ws;
  size_t off = 0;
  float* x   = ws + off; off += (size_t)MROWS * DD;   // residual stream
  float* hn  = ws + off; off += (size_t)MROWS * DD;   // normed activations
  float* qb  = ws + off; off += (size_t)MROWS * DD;
  float* kb  = ws + off; off += (size_t)MROWS * NKV * HD;
  float* vb  = ws + off; off += (size_t)MROWS * NKV * HD;
  float* ob  = ws + off; off += (size_t)MROWS * DD;
  float* gb  = ws + off; off += (size_t)MROWS * FF;
  float* ub  = ws + off; off += (size_t)MROWS * FF;
  float* fb  = ws + off; off += (size_t)MROWS * FF;
  float* h1  = ws + off; off += (size_t)16 * 4 * DD;  // padded to 16 rows
  float* t1  = ws + off; off += (size_t)MROWS * RR;
  float* logits = ws + off; off += (size_t)MROWS * VV;

  set_zero_kernel<<<1, 1, 0, stream>>>(out);

  // conditioner: h1 = gelu(s_bt@w1+b1) padded; P = h1@w2+b2 into x[:, :16, :]
  cond_h1_kernel<<<(16 * 4 * DD + 255) / 256, 256, 0, stream>>>(s_bt, w1, b1, h1);
  gemm_f32_kernel<NTILE><<<gemm_grid(16, PREFIX * DD), 256, 0, stream>>>(
      h1, 16, 4 * DD, w2, PREFIX * DD, b2, PREFIX * DD, x, TT * DD, BB);
  embed_gather_kernel<<<dim3(PROMPT + LAB, BB), 256, 0, stream>>>(
      embed, prompt_ids, label_ids, x);

  for (int l = 0; l < NL; ++l) {
    const int*   qc = q_codes + (size_t)l * DD * DD;
    const float* qa = q_absmax + (size_t)l * (DD * DD / 64);
    const int*   kc = k_codes + (size_t)l * NKV * HD * DD;
    const float* ka = k_absmax + (size_t)l * (NKV * HD * DD / 64);
    const int*   vc = v_codes + (size_t)l * NKV * HD * DD;
    const float* va = v_absmax + (size_t)l * (NKV * HD * DD / 64);
    const int*   oc = o_codes + (size_t)l * DD * DD;
    const float* oa = o_absmax + (size_t)l * (DD * DD / 64);
    const int*   gc = g_codes + (size_t)l * FF * DD;
    const float* ga = g_absmax + (size_t)l * (FF * DD / 64);
    const int*   uc = u_codes + (size_t)l * FF * DD;
    const float* ua = u_absmax + (size_t)l * (FF * DD / 64);
    const int*   dc = d_codes + (size_t)l * DD * FF;
    const float* da = d_absmax + (size_t)l * (DD * FF / 64);

    rmsnorm_kernel<<<MROWS, 256, 0, stream>>>(x, attn_norm + (size_t)l * DD, hn);

    gemm_nf4_kernel<NTILE><<<gemm_grid(MROWS, DD), 256, 0, stream>>>(
        hn, MROWS, DD, qc, qa, DD, qb, DD, 0);
    gemm_nf4_kernel<NTILE><<<gemm_grid(MROWS, NKV * HD), 256, 0, stream>>>(
        hn, MROWS, DD, kc, ka, NKV * HD, kb, NKV * HD, 0);
    gemm_nf4_kernel<NTILE><<<gemm_grid(MROWS, NKV * HD), 256, 0, stream>>>(
        hn, MROWS, DD, vc, va, NKV * HD, vb, NKV * HD, 0);

    lora_a_kernel<<<(MROWS * RR + 255) / 256, 256, 0, stream>>>(
        hn, loraA + (size_t)l * RR * DD, t1);
    lora_b_add_kernel<<<(MROWS * DD + 255) / 256, 256, 0, stream>>>(
        t1, loraB + (size_t)l * DD * RR, qb);

    rope_kernel<<<(MROWS * NH * 32 + 255) / 256, 256, 0, stream>>>(qb, NH);
    rope_kernel<<<(MROWS * NKV * 32 + 255) / 256, 256, 0, stream>>>(kb, NKV);

    attn_kernel<<<dim3(BB, NH), 128, 0, stream>>>(qb, kb, vb, ob);

    gemm_nf4_kernel<NTILE><<<gemm_grid(MROWS, DD), 256, 0, stream>>>(
        ob, MROWS, DD, oc, oa, DD, x, DD, 1);   // x += o @ Wo^T

    rmsnorm_kernel<<<MROWS, 256, 0, stream>>>(x, mlp_norm + (size_t)l * DD, hn);
    gemm_nf4_kernel<NTILE><<<gemm_grid(MROWS, FF), 256, 0, stream>>>(
        hn, MROWS, DD, gc, ga, FF, gb, FF, 0);
    gemm_nf4_kernel<NTILE><<<gemm_grid(MROWS, FF), 256, 0, stream>>>(
        hn, MROWS, DD, uc, ua, FF, ub, FF, 0);
    silu_mul_kernel<<<(MROWS * FF + 255) / 256, 256, 0, stream>>>(gb, ub, fb);
    gemm_nf4_kernel<NTILE><<<gemm_grid(MROWS, DD), 256, 0, stream>>>(
        fb, MROWS, FF, dc, da, DD, x, DD, 1);   // x += ff @ Wd^T
  }

  rmsnorm_kernel<<<MROWS, 256, 0, stream>>>(x, final_norm, hn);
  gemm_f32_kernel<NTILE><<<gemm_grid(MROWS, VV), 256, 0, stream>>>(
      hn, MROWS, DD, lm_head, VV, nullptr, VV, logits, VV, MROWS);
  nll_kernel<<<dim3(LAB, BB), 256, 0, stream>>>(logits, label_ids, out);
}
